// DenseLayer_76682346103544
// MI455X (gfx1250) — matrix operand order repaired and checked by reading, not compiled, not run on hardware
//
#include <hip/hip_runtime.h>

typedef __attribute__((ext_vector_type(16))) _Float16 v16h;
typedef __attribute__((ext_vector_type(8)))  _Float16 v8h;
typedef __attribute__((ext_vector_type(8)))  float    v8f;
typedef __attribute__((ext_vector_type(4)))  float    f4;

#define B_SZ 32
#define T_SZ 1000
#define I_SZ 256
#define H_SZ 512
#define M_SZ (B_SZ * T_SZ)          // 32000 rows of x2d

// exp(-1/20) and 1 - exp(-1/20); tau_mem == tau_trace == 20
#define DECAY_MEM 0.95122942450071400910f
#define ONE_M_DM  0.04877057549928599090f
#define DECAY_TR  0.95122942450071400910f
#define ONE_M_DT  0.04877057549928599090f

// ---------------------------------------------------------------------------
// Phase 0: one-shot fp32 -> f16 conversion of x and W (streaming, 8 elems/thread)
// ---------------------------------------------------------------------------
__global__ __launch_bounds__(256)
void snn_cvt_f16(const float* __restrict__ x, const float* __restrict__ W,
                 _Float16* __restrict__ xh, _Float16* __restrict__ wh) {
    const size_t id = (size_t)blockIdx.x * 256 + threadIdx.x;
    const size_t nx = (size_t)M_SZ * I_SZ / 8;   // 1,024,000 groups of 8
    const size_t nw = (size_t)H_SZ * I_SZ / 8;   //    16,384 groups of 8
    if (id < nx) {
        const f4* p = (const f4*)x + id * 2;
        f4 a = p[0], b = p[1];
        v8h o;
        #pragma unroll
        for (int e = 0; e < 4; ++e) { o[e] = (_Float16)a[e]; o[e + 4] = (_Float16)b[e]; }
        ((v8h*)xh)[id] = o;
    } else if (id < nx + nw) {
        const size_t j = id - nx;
        const f4* p = (const f4*)W + j * 2;
        f4 a = p[0], b = p[1];
        v8h o;
        #pragma unroll
        for (int e = 0; e < 4; ++e) { o[e] = (_Float16)a[e]; o[e + 4] = (_Float16)b[e]; }
        ((v8h*)wh)[j] = o;
    }
}

// Build a 16-wide f16 fragment from two b128 (8 x f16) loads.
__device__ __forceinline__ v16h frag16(const _Float16* lo, const _Float16* hi) {
    v8h a = *(const v8h*)lo;
    v8h b = *(const v8h*)hi;
    return __builtin_shufflevector(a, b, 0, 1, 2, 3, 4, 5, 6, 7,
                                         8, 9, 10, 11, 12, 13, 14, 15);
}

// ---------------------------------------------------------------------------
// Phase 1 (fast path): 32x32 output per wave = 2x2 WMMA tiles, f16 inputs.
// Per K-step of 32: 8 global_load_b128 + 4 v_wmma; A shared across n-tiles,
// B shared across m-tiles -> 4x less L2 traffic per output than 16x16/wave.
// ---------------------------------------------------------------------------
__global__ __launch_bounds__(256)
void snn_gemm_f16_wmma_32x32(const _Float16* __restrict__ xh,
                             const _Float16* __restrict__ wh,
                             float* __restrict__ cur) {
    const int lane = threadIdx.x & 31;
    const int wave = threadIdx.x >> 5;
    const int g      = blockIdx.x * 8 + wave;   // tile id, 0..15999
    const int tile_m = g >> 4;                  // / (H/32 == 16): 0..999
    const int tile_n = g & 15;

    const int l15  = lane & 15;
    const int hi   = lane >> 4;                 // lane half: 0 or 1
    const int aOff = hi * 16;                   // A K-offset (f16 elems)
    const int bOff = hi * 16;                   // B K-offset (f16 elems)

    // A rows (M) for the two m-subtiles; B rows of W (= columns N) for n-subtiles
    const _Float16* __restrict__ a0 = xh + (size_t)(tile_m * 32 + l15) * I_SZ;
    const _Float16* __restrict__ a1 = a0 + (size_t)16 * I_SZ;
    const _Float16* __restrict__ b0 = wh + (size_t)(tile_n * 32 + l15) * I_SZ;
    const _Float16* __restrict__ b1 = b0 + (size_t)16 * I_SZ;

    v8f acc00 = {}, acc01 = {}, acc10 = {}, acc11 = {};
    #pragma unroll
    for (int k0 = 0; k0 < I_SZ; k0 += 32) {
        v16h A0 = frag16(a0 + k0 + aOff, a0 + k0 + 8 + aOff);
        v16h A1 = frag16(a1 + k0 + aOff, a1 + k0 + 8 + aOff);
        v16h B0 = frag16(b0 + k0 + bOff, b0 + k0 + 8 + bOff);
        v16h B1 = frag16(b1 + k0 + bOff, b1 + k0 + 8 + bOff);
        acc00 = __builtin_amdgcn_wmma_f32_16x16x32_f16(false, A0, false, B0,
                                                       (short)0, acc00, false, false);
        acc01 = __builtin_amdgcn_wmma_f32_16x16x32_f16(false, A0, false, B1,
                                                       (short)0, acc01, false, false);
        acc10 = __builtin_amdgcn_wmma_f32_16x16x32_f16(false, A1, false, B0,
                                                       (short)0, acc10, false, false);
        acc11 = __builtin_amdgcn_wmma_f32_16x16x32_f16(false, A1, false, B1,
                                                       (short)0, acc11, false, false);
    }

    // D layout: VGPR r holds (M = r + hi*8, N = l15)
    float* __restrict__ op =
        cur + (size_t)(tile_m * 32 + hi * 8) * H_SZ + tile_n * 32 + l15;
    float* __restrict__ oq = op + (size_t)16 * H_SZ;
    #pragma unroll
    for (int r = 0; r < 8; ++r) {
        op[(size_t)r * H_SZ]      = acc00[r];
        op[(size_t)r * H_SZ + 16] = acc01[r];
        oq[(size_t)r * H_SZ]      = acc10[r];
        oq[(size_t)r * H_SZ + 16] = acc11[r];
    }
}

// ---------------------------------------------------------------------------
// Phase 1 (fallback if workspace too small for f16 staging): fp32 in,
// convert-in-kernel, 16x16 per wave (round-1 kernel).
// ---------------------------------------------------------------------------
__global__ __launch_bounds__(256)
void snn_gemm_f16_wmma_cvt(const float* __restrict__ x,
                           const float* __restrict__ W,
                           float* __restrict__ cur) {
    const int lane = threadIdx.x & 31;
    const int wave = threadIdx.x >> 5;
    const int g      = blockIdx.x * 8 + wave;   // tile id, 0..63999
    const int tile_m = g >> 5;
    const int tile_n = g & 31;

    const int l15   = lane & 15;
    const int hi    = lane >> 4;
    const int aBase = hi * 16;
    const int bBase = hi * 16;

    const float* __restrict__ xrow = x + (size_t)(tile_m * 16 + l15) * I_SZ;
    const float* __restrict__ wrow = W + (size_t)(tile_n * 16 + l15) * I_SZ;

    v8f acc = {};
    #pragma unroll
    for (int k0 = 0; k0 < I_SZ; k0 += 32) {
        const f4* pa0 = (const f4*)(xrow + k0 + aBase);
        const f4* pa1 = (const f4*)(xrow + k0 + 8 + aBase);
        const f4* pb  = (const f4*)(wrow + k0 + bBase);
        f4 a0 = pa0[0], a1 = pa0[1];
        f4 a2 = pa1[0], a3 = pa1[1];
        f4 b0 = pb[0],  b1 = pb[1], b2 = pb[2], b3 = pb[3];

        v16h A, Bm;
        #pragma unroll
        for (int e = 0; e < 4; ++e) {
            A[e]      = (_Float16)a0[e];
            A[e + 4]  = (_Float16)a1[e];
            A[e + 8]  = (_Float16)a2[e];
            A[e + 12] = (_Float16)a3[e];
            Bm[e]      = (_Float16)b0[e];
            Bm[e + 4]  = (_Float16)b1[e];
            Bm[e + 8]  = (_Float16)b2[e];
            Bm[e + 12] = (_Float16)b3[e];
        }
        acc = __builtin_amdgcn_wmma_f32_16x16x32_f16(false, A, false, Bm,
                                                     (short)0, acc, false, false);
    }

    float* __restrict__ op =
        cur + (size_t)(tile_m * 16 + hi * 8) * H_SZ + tile_n * 16 + l15;
    #pragma unroll
    for (int r = 0; r < 8; ++r)
        op[(size_t)r * H_SZ] = acc[r];
}

// ---------------------------------------------------------------------------
// Phase 2: sequential LIF scan over T per (b,h). Coalesced along h.
// ---------------------------------------------------------------------------
__global__ __launch_bounds__(256)
void snn_scan(const float* __restrict__ cur,
              float* __restrict__ zout,
              float* __restrict__ trout) {
    const int idx = blockIdx.x * 256 + threadIdx.x;  // 0 .. B*H-1
    const int b = idx >> 9;                          // / 512
    const int h = idx & (H_SZ - 1);

    const float* __restrict__ ip = cur   + (size_t)b * T_SZ * H_SZ + h;
    float* __restrict__ zp       = zout  + (size_t)b * T_SZ * H_SZ + h;
    float* __restrict__ tp       = trout + (size_t)b * T_SZ * H_SZ + h;

    float v = 0.0f, z = 0.0f, tr = 0.0f;
    #pragma unroll 4
    for (int t = 0; t < T_SZ; ++t) {
        float inp = __builtin_nontemporal_load(ip);  ip += H_SZ;
        v  = DECAY_MEM * v * (1.0f - z) + ONE_M_DM * inp;
        z  = (v > 1.0f) ? 1.0f : 0.0f;               // THR == 1
        tr = DECAY_TR * tr + ONE_M_DT * z;
        __builtin_nontemporal_store(z,  zp);  zp += H_SZ;
        __builtin_nontemporal_store(tr, tp);  tp += H_SZ;
    }
}

// ---------------------------------------------------------------------------
extern "C" void kernel_launch(void* const* d_in, const int* in_sizes, int n_in,
                              void* d_out, int out_size, void* d_ws, size_t ws_size,
                              hipStream_t stream) {
    const float* x = (const float*)d_in[0];   // [B,T,I] = [32,1000,256]
    const float* W = (const float*)d_in[1];   // [H,I]   = [512,256]

    const size_t curBytes = (size_t)M_SZ * H_SZ * sizeof(float);       // 65.5 MB
    const size_t xhBytes  = (size_t)M_SZ * I_SZ * sizeof(_Float16);    // 16.4 MB
    const size_t whBytes  = (size_t)H_SZ * I_SZ * sizeof(_Float16);    // 0.26 MB

    float* cur   = (float*)d_ws;
    float* zout  = (float*)d_out;                                      // out_seq
    float* trout = zout + (size_t)B_SZ * T_SZ * H_SZ;                  // trace_seq

    if (ws_size >= curBytes + xhBytes + whBytes) {
        // Fast path: stage f16 copies of x and W, then 32x32-per-wave WMMA GEMM.
        _Float16* xh = (_Float16*)((char*)d_ws + curBytes);
        _Float16* wh = (_Float16*)((char*)d_ws + curBytes + xhBytes);

        const int nCvt = (M_SZ * I_SZ + H_SZ * I_SZ) / 8;              // 1,040,384
        snn_cvt_f16<<<(nCvt + 255) / 256, 256, 0, stream>>>(x, W, xh, wh);

        // 1000 m-tiles * 16 n-tiles = 16000 wave-tiles, 8 waves / block
        snn_gemm_f16_wmma_32x32<<<(M_SZ / 32) * (H_SZ / 32) / 8, 256, 0, stream>>>(
            xh, wh, cur);
    } else {
        // Fallback: fp32 loads with in-kernel conversion, 16x16 per wave.
        snn_gemm_f16_wmma_cvt<<<(M_SZ / 16) * (H_SZ / 16) / 8, 256, 0, stream>>>(
            x, W, cur);
    }

    // one thread per (b,h): 16384 threads
    snn_scan<<<(B_SZ * H_SZ) / 256, 256, 0, stream>>>(cur, zout, trout);
}
